// MS_SSIM_2138893713911
// MI455X (gfx1250) — matrix operand order repaired and checked by reading, not compiled, not run on hardware
//
#include <hip/hip_runtime.h>
#include <hip/hip_bf16.h>

typedef _Float16 v16h __attribute__((ext_vector_type(16)));
typedef _Float16 h8  __attribute__((ext_vector_type(8)));
typedef float    v8f __attribute__((ext_vector_type(8)));

#define MAP_STRIDE 40   // halves per LDS map row (80B, keeps 16B alignment for 8-half chunks)
#define NPLANES    24   // N*C = 8*3

__device__ __forceinline__ v8f wmma_f16(v16h a, v16h b, v8f c) {
    // D = A(16x32 f16) * B(32x16 f16) + C(16x16 f32)
    return __builtin_amdgcn_wmma_f32_16x16x32_f16(false, a, false, b, (short)0, c, false, false);
}

// One wave per 16x16 output tile. Fused separable 11-tap Gaussian conv of the
// five SSIM maps via banded-matrix WMMA, then per-tile cs/ss reduction.
__global__ void __launch_bounds__(32)
ssim_tile_kernel(const float* __restrict__ X, const float* __restrict__ Y,
                 const float* __restrict__ G, float* __restrict__ accum,
                 int H, int W, int lastScale)
{
    __shared__ __align__(16) _Float16 maps[5][32 * MAP_STRIDE]; // x, y, xx, yy, xy (f16)
    __shared__ __align__(32) _Float16 Sld[16 * 32];             // stage-1 result, column-major [col][row]
    __shared__ float gs[16];                                     // gaussian taps

    const int lane  = threadIdx.x;
    const int plane = blockIdx.z;
    const int r0 = blockIdx.y << 4;
    const int c0 = blockIdx.x << 4;
    const int outH = H - 10, outW = W - 10;

    if (lane < 11) gs[lane] = G[lane];

    // ---- load 32x32 input tile (edge-clamped), build the 5 maps in LDS as f16 ----
    const size_t planeOff = (size_t)plane * H * W;
    int gr = r0 + lane; if (gr > H - 1) gr = H - 1;
    const float* xr = X + planeOff + (size_t)gr * W;
    const float* yr = Y + planeOff + (size_t)gr * W;
    _Float16* m0 = &maps[0][lane * MAP_STRIDE];
    _Float16* m1 = &maps[1][lane * MAP_STRIDE];
    _Float16* m2 = &maps[2][lane * MAP_STRIDE];
    _Float16* m3 = &maps[3][lane * MAP_STRIDE];
    _Float16* m4 = &maps[4][lane * MAP_STRIDE];

    if (c0 + 32 <= W) {
        #pragma unroll
        for (int j = 0; j < 8; ++j) {
            float4 a = *(const float4*)(xr + c0 + 4 * j);
            float4 b = *(const float4*)(yr + c0 + 4 * j);
            float xa[4] = {a.x, a.y, a.z, a.w};
            float yb[4] = {b.x, b.y, b.z, b.w};
            #pragma unroll
            for (int t = 0; t < 4; ++t) {
                int c = 4 * j + t;
                float xv = xa[t], yv = yb[t];
                m0[c] = (_Float16)xv;          m1[c] = (_Float16)yv;
                m2[c] = (_Float16)(xv * xv);   m3[c] = (_Float16)(yv * yv);
                m4[c] = (_Float16)(xv * yv);
            }
        }
    } else {
        #pragma unroll
        for (int j = 0; j < 32; ++j) {
            int gc = c0 + j; if (gc > W - 1) gc = W - 1;
            float xv = xr[gc], yv = yr[gc];
            m0[j] = (_Float16)xv;          m1[j] = (_Float16)yv;
            m2[j] = (_Float16)(xv * xv);   m3[j] = (_Float16)(yv * yv);
            m4[j] = (_Float16)(xv * yv);
        }
    }
    __syncthreads();

    const int al = lane & 15;   // M (A) / N (B,D) coordinate
    const int hi = lane >> 4;   // lane half

    // ---- banded Gaussian operands ----
    // b1: stage-1 B (32x16):  B[k][n] = g[k-n]   (conv along columns)
    // a2: stage-2 A (16x32):  A[m][k] = g[k-m]   (conv along rows)
    v16h b1, a2;
    #pragma unroll
    for (int e = 0; e < 16; ++e) {
        int k1 = (e < 8) ? (hi * 8 + e) : (16 + hi * 8 + (e - 8)); // B layout K (table order, matches aT/aB)
        int d1 = k1 - al;
        bool ok1 = (unsigned)d1 < 11u;
        float t1 = gs[ok1 ? d1 : 0];
        b1[e] = ok1 ? (_Float16)t1 : (_Float16)0.0f;
        int k2 = hi * 16 + e;                       // A layout: K = 16*half + elem (matches bS)
        int d2 = k2 - al;
        bool ok2 = (unsigned)d2 < 11u;
        float t2 = gs[ok2 ? d2 : 0];
        a2[e] = ok2 ? (_Float16)t2 : (_Float16)0.0f;
    }

    // ---- per-map: S = In*B (2 WMMA), relayout via LDS, Out = A*S (1 WMMA) ----
    v8f accv[5];
    #pragma unroll
    for (int m = 0; m < 5; ++m) {
        const _Float16* mb = &maps[m][0];
        const _Float16* pT = mb + al * MAP_STRIDE + hi * 8;        // rows 0..15
        const _Float16* pB = mb + (16 + al) * MAP_STRIDE + hi * 8; // rows 16..31
        h8 tlo = *(const h8*)pT;  h8 thi = *(const h8*)(pT + 16);
        h8 blo = *(const h8*)pB;  h8 bhi = *(const h8*)(pB + 16);
        v16h aT = __builtin_shufflevector(tlo, thi, 0,1,2,3,4,5,6,7,8,9,10,11,12,13,14,15);
        v16h aB = __builtin_shufflevector(blo, bhi, 0,1,2,3,4,5,6,7,8,9,10,11,12,13,14,15);

        v8f z;
        #pragma unroll
        for (int e = 0; e < 8; ++e) z[e] = 0.0f;

        v8f sT = wmma_f16(aT, b1, z);   // S rows  0..15
        v8f sB = wmma_f16(aB, b1, z);   // S rows 16..31

        h8 hT, hB;
        #pragma unroll
        for (int e = 0; e < 8; ++e) { hT[e] = (_Float16)sT[e]; hB[e] = (_Float16)sB[e]; }

        __syncthreads();                 // protect Sld reuse across maps
        _Float16* sc = &Sld[al * 32];    // column-major: col = al, stride 32 rows
        *(h8*)(sc + hi * 8)      = hT;   // D layout: lane half picks rows {0..7|8..15}
        *(h8*)(sc + 16 + hi * 8) = hB;   //                       rows {16..23|24..31}
        __syncthreads();

        // B-operand load: N = al, K = 16*hi + 0..15 -> contiguous in column-major Sld
        v16h bS = *(const v16h*)(&Sld[al * 32 + hi * 16]);
        accv[m] = wmma_f16(a2, bS, z);
    }

    // ---- per-element cs/ss and tile reduction (D layout: N=al, M=v+8*hi) ----
    const float c1 = 0.01f * 0.01f, c2 = 0.03f * 0.03f;
    float sumv = 0.0f;
    const int M0 = hi * 8;
    #pragma unroll
    for (int v = 0; v < 8; ++v) {
        int orow = r0 + M0 + v;
        int ocol = c0 + al;
        if (orow < outH && ocol < outW) {
            float mx = accv[0][v], my = accv[1][v];
            float exx = accv[2][v], eyy = accv[3][v], exy = accv[4][v];
            float mxx = mx * mx, myy = my * my, mxy = mx * my;
            float cs = (2.0f * (exy - mxy) + c2) / ((exx - mxx) + (eyy - myy) + c2);
            float val = lastScale ? (2.0f * mxy + c1) / (mxx + myy + c1) * cs : cs;
            sumv += val;
        }
    }
    #pragma unroll
    for (int off = 16; off; off >>= 1) sumv += __shfl_down(sumv, off, 32);
    if (lane == 0) atomicAdd(&accum[plane], sumv);
}

__global__ void pool_kernel(const float* __restrict__ src, float* __restrict__ dst,
                            int D, int total)
{
    int i = blockIdx.x * blockDim.x + threadIdx.x;
    if (i >= total) return;
    int Dh = D >> 1;
    int w = i % Dh; int t = i / Dh; int h = t % Dh; int p = t / Dh;
    const float* sp = src + ((size_t)p * D + 2 * h) * D + 2 * w;
    dst[i] = 0.25f * (sp[0] + sp[1] + sp[D] + sp[D + 1]);
}

__global__ void zero_kernel(float* p, int n)
{
    int i = blockIdx.x * blockDim.x + threadIdx.x;
    if (i < n) p[i] = 0.0f;
}

__global__ void __launch_bounds__(32)
finalize_kernel(const float* __restrict__ acc, const float* __restrict__ w,
                float* __restrict__ out)
{
    int l = threadIdx.x;
    float t = 0.0f;
    if (l < NPLANES) {
        const float cnt[5] = {502.f * 502.f, 246.f * 246.f, 118.f * 118.f,
                              54.f * 54.f, 22.f * 22.f};
        t = 1.0f;
        #pragma unroll
        for (int s = 0; s < 5; ++s) {
            float m = acc[s * NPLANES + l] / cnt[s];
            m = fmaxf(m, 0.0f);
            t *= powf(m, w[s]);
        }
    }
    #pragma unroll
    for (int off = 16; off; off >>= 1) t += __shfl_down(t, off, 32);
    if (l == 0) out[0] = t / (float)NPLANES;
}

extern "C" void kernel_launch(void* const* d_in, const int* in_sizes, int n_in,
                              void* d_out, int out_size, void* d_ws, size_t ws_size,
                              hipStream_t stream)
{
    (void)in_sizes; (void)n_in; (void)out_size; (void)ws_size;

    const float* X0 = (const float*)d_in[0];   // (8,3,512,512)
    const float* Y0 = (const float*)d_in[1];
    const float* G  = (const float*)d_in[2];   // (3,1,11) -> first 11 taps
    const float* Wt = (const float*)d_in[3];   // (5,)
    float* out = (float*)d_out;
    float* ws  = (float*)d_ws;

    float* acc = ws;           // 5*24 accumulators (padded to 128)
    float* buf = ws + 128;     // image pyramid

    static const int dims[5] = {512, 256, 128, 64, 32};
    const float* xs[5]; const float* ys[5];
    float* mx[5]; float* my[5];
    xs[0] = X0; ys[0] = Y0;
    size_t off = 0;
    for (int s = 1; s < 5; ++s) {
        size_t n = (size_t)NPLANES * dims[s] * dims[s];
        mx[s] = buf + off; off += n;
        my[s] = buf + off; off += n;
        xs[s] = mx[s]; ys[s] = my[s];
    }

    zero_kernel<<<1, 128, 0, stream>>>(acc, 128);

    for (int s = 0; s < 5; ++s) {
        if (s > 0) {
            int D = dims[s - 1];
            int total = NPLANES * (D / 2) * (D / 2);
            int blk = 256, grd = (total + blk - 1) / blk;
            pool_kernel<<<grd, blk, 0, stream>>>(xs[s - 1], mx[s], D, total);
            pool_kernel<<<grd, blk, 0, stream>>>(ys[s - 1], my[s], D, total);
        }
        int d = dims[s];
        int od = d - 10;
        int tiles = (od + 15) / 16;
        dim3 grid(tiles, tiles, NPLANES);
        ssim_tile_kernel<<<grid, 32, 0, stream>>>(xs[s], ys[s], G,
                                                  acc + s * NPLANES, d, d,
                                                  (s == 4) ? 1 : 0);
    }

    finalize_kernel<<<1, 32, 0, stream>>>(acc, Wt, out);
}
